// RegionSelectionAttention_2963527434300
// MI455X (gfx1250) — matrix operand order repaired and checked by reading, not compiled, not run on hardware
//
#include <hip/hip_runtime.h>
#include <hip/hip_bf16.h>
#include <math.h>

// ================= CDNA5 (gfx1250, wave32) WMMA plumbing =================
typedef __attribute__((ext_vector_type(16))) _Float16 v16h;
typedef __attribute__((ext_vector_type(8)))  float    v8f;

#define DEVINL static __device__ __forceinline__

DEVINL v8f wmma16(v16h a, v16h b, v8f c) {
  // D = A(16x32,f16) * B(32x16,f16) + C(16x16,f32)
  return __builtin_amdgcn_wmma_f32_16x16x32_f16(false, a, false, b, (short)0, c,
                                                false, false);
}

// A fragment (16 rows x 32 K) from row-major f16 rows at `base`, leading dim ld.
// lanes 0-15 row=lane, K {0..15}; lanes 16-31 row=lane-16,
// K {16..31}.
DEVINL v16h frag_a_f16(const _Float16* base, int ld, int lane, int k0) {
  int row = lane & 15, sel = lane >> 4;
  const _Float16* p = base + (size_t)row * ld + k0 + sel * 16;
  v16h f;
#pragma unroll
  for (int e = 0; e < 16; ++e) f[e] = p[e];
  return f;
}

// B fragment (32 K x 16 N) from row-major f16 B[K][N] (strided per lane).
DEVINL v16h frag_b_rm_f16(const _Float16* base, int ld, int lane, int k0) {
  int n = lane & 15, sel = lane >> 4;
  const _Float16* p = base + (size_t)(k0 + sel * 16) * ld + n;
  v16h f;
#pragma unroll
  for (int e = 0; e < 16; ++e) f[e] = p[(size_t)e * ld];
  return f;
}

// B fragment where memory holds W[N][K] row-major (B = W^T): contiguous loads.
DEVINL v16h frag_b_nm_f16(const _Float16* base, int ld, int lane, int k0) {
  int n = lane & 15, sel = lane >> 4;
  const _Float16* p = base + (size_t)n * ld + k0 + sel * 16;
  v16h f;
#pragma unroll
  for (int e = 0; e < 16; ++e) f[e] = p[e];
  return f;
}

// reductions across a 16-lane half (xor masks 1..8 never cross bit4)
DEVINL float hmax16(float v) {
#pragma unroll
  for (int m = 1; m <= 8; m <<= 1) v = fmaxf(v, __shfl_xor(v, m, 32));
  return v;
}
DEVINL float hsum16(float v) {
#pragma unroll
  for (int m = 1; m <= 8; m <<= 1) v += __shfl_xor(v, m, 32);
  return v;
}

// ================= problem constants =================
// B=8, C=256, H=W=64, nh=4, hd=64, h=w=32, hw=1024, G=B*nh=32, Ktop=256

// ---- 0a) x: NCHW f32 -> NHWC f16 (one-time; makes down-conv A frags contiguous) ----
__global__ void k_x_to_nhwc(const float* __restrict__ x, _Float16* __restrict__ xh) {
  size_t i = (size_t)blockIdx.x * blockDim.x + threadIdx.x;
  if (i >= (size_t)8 * 256 * 4096) return;
  int X = (int)(i & 63), Y = (int)((i >> 6) & 63);
  int c = (int)((i >> 12) & 255), b = (int)(i >> 20);
  xh[(((size_t)b * 64 + Y) * 64 + X) * 256 + c] = (_Float16)x[i];
}

// ---- 0b) down_w [n][cin][ky][kx] f32 -> [n][tap][cin] f16 (kk = tap*256+cin) ----
__global__ void k_w_down_pack(const float* __restrict__ w, _Float16* __restrict__ wd) {
  int i = blockIdx.x * blockDim.x + threadIdx.x;
  if (i >= 1048576) return;
  int tap = i & 15, cin = (i >> 4) & 255, n = i >> 12;
  wd[(size_t)n * 4096 + (size_t)tap * 256 + cin] = (_Float16)w[i];
}

// ---- 0c) up_w [cin][n][ky][kx] f32 -> [n][ky*4+kx][cin] f16 ----
__global__ void k_w_up_pack(const float* __restrict__ w, _Float16* __restrict__ wu) {
  int i = blockIdx.x * blockDim.x + threadIdx.x;
  if (i >= 1048576) return;
  int t = i & 15, n = (i >> 4) & 255, cin = i >> 12;
  wu[(size_t)n * 4096 + (size_t)t * 256 + cin] = (_Float16)w[i];
}

// ---- 0d) generic f32 -> f16 copy (qkv weights [N][K], pw_w [n][cin]) ----
__global__ void k_w_cvt(const float* __restrict__ w, _Float16* __restrict__ o, int nel) {
  int i = blockIdx.x * blockDim.x + threadIdx.x;
  if (i < nel) o[i] = (_Float16)w[i];
}

// ---- 1) down conv 4x4 s2 p1 as implicit GEMM: M=8192(b,p) N=256 K=4096 ----
// writes f16 tokens tok[g=b*4+head][p][d]
__global__ __launch_bounds__(32)
void k_down(const _Float16* __restrict__ xh, const _Float16* __restrict__ wd,
            const float* __restrict__ bias, _Float16* __restrict__ tok) {
  const int lane = threadIdx.x, sel = lane >> 4, ln = lane & 15;
  const int m0 = blockIdx.x * 32, n0 = blockIdx.y * 32;
  v8f acc[2][2] = {};
  int ob[2], oy[2], ox[2];
#pragma unroll
  for (int mt = 0; mt < 2; ++mt) {
    int m = m0 + mt * 16 + ln;
    ob[mt] = m >> 10; int p = m & 1023; oy[mt] = p >> 5; ox[mt] = p & 31;
  }
  for (int k0 = 0; k0 < 4096; k0 += 32) {
    int tap = k0 >> 8;                 // kk = tap*256 + cin, constant per 32-chunk
    int ky = tap >> 2, kx = tap & 3;
    v16h a[2];
#pragma unroll
    for (int mt = 0; mt < 2; ++mt) {
      int iy = oy[mt] * 2 + ky - 1, ix = ox[mt] * 2 + kx - 1;
      bool ok = (unsigned)iy < 64u && (unsigned)ix < 64u;
      const _Float16* ap = xh + (((size_t)ob[mt] * 64 + iy) * 64 + ix) * 256 +
                           (k0 & 255) + sel * 16;
      if (ok) {
        __builtin_prefetch(ap + 256, 0, 1);
#pragma unroll
        for (int e = 0; e < 16; ++e) a[mt][e] = ap[e];
      } else {
#pragma unroll
        for (int e = 0; e < 16; ++e) a[mt][e] = (_Float16)0.f;
      }
    }
    v16h bf[2];
#pragma unroll
    for (int nt = 0; nt < 2; ++nt)
      bf[nt] = frag_b_nm_f16(wd + (size_t)(n0 + nt * 16) * 4096, 4096, lane, k0);
#pragma unroll
    for (int mt = 0; mt < 2; ++mt)
#pragma unroll
      for (int nt = 0; nt < 2; ++nt)
        acc[mt][nt] = wmma16(a[mt], bf[nt], acc[mt][nt]);
  }
#pragma unroll
  for (int mt = 0; mt < 2; ++mt)
#pragma unroll
    for (int nt = 0; nt < 2; ++nt)
#pragma unroll
      for (int r8 = 0; r8 < 8; ++r8) {
        int m = m0 + mt * 16 + sel * 8 + r8;
        int n = n0 + nt * 16 + ln;
        float v = acc[mt][nt][r8] + bias[n];
        int b = m >> 10, p = m & 1023, head = n >> 6, d = n & 63;
        tok[(((size_t)(b * 4 + head)) * 1024 + p) * 64 + d] = (_Float16)v;
      }
}

// ---- 2) qkv projection: M=G*1024, N=192, K=64 (shared for both attentions) ----
__global__ __launch_bounds__(32)
void k_qkv(const _Float16* __restrict__ tok, const _Float16* __restrict__ w,
           const float* __restrict__ bias, _Float16* __restrict__ qb,
           _Float16* __restrict__ kb, _Float16* __restrict__ vb) {
  const int lane = threadIdx.x, sel = lane >> 4, ln = lane & 15;
  const int m0 = blockIdx.x * 32, n0 = blockIdx.y * 32;
  v8f acc[2][2] = {};
  for (int k0 = 0; k0 < 64; k0 += 32) {
    v16h a[2], bf[2];
#pragma unroll
    for (int mt = 0; mt < 2; ++mt)
      a[mt] = frag_a_f16(tok + (size_t)(m0 + mt * 16) * 64, 64, lane, k0);
#pragma unroll
    for (int nt = 0; nt < 2; ++nt)
      bf[nt] = frag_b_nm_f16(w + (size_t)(n0 + nt * 16) * 64, 64, lane, k0);
#pragma unroll
    for (int mt = 0; mt < 2; ++mt)
#pragma unroll
      for (int nt = 0; nt < 2; ++nt)
        acc[mt][nt] = wmma16(a[mt], bf[nt], acc[mt][nt]);
  }
#pragma unroll
  for (int mt = 0; mt < 2; ++mt)
#pragma unroll
    for (int nt = 0; nt < 2; ++nt)
#pragma unroll
      for (int r8 = 0; r8 < 8; ++r8) {
        int m = m0 + mt * 16 + sel * 8 + r8;
        int n = n0 + nt * 16 + ln;
        float v = acc[mt][nt][r8] + bias[n];
        int g = m >> 10, p = m & 1023, which = n >> 6, d = n & 63;
        _Float16* dst = (which == 0) ? qb : (which == 1) ? kb : vb;
        dst[((size_t)g * 1024 + p) * 64 + d] = (_Float16)v;
      }
}

// ---- 3) attention pass1: S = K.Q^T row-max / row-sumexp (online) ----
__global__ __launch_bounds__(32)
void k_attn_pass1(const _Float16* __restrict__ kb, const _Float16* __restrict__ qb,
                  float* __restrict__ rowmax, float* __restrict__ rowsum) {
  const int lane = threadIdx.x, sel = lane >> 4, ln = lane & 15;
  const int g = blockIdx.y, i0 = blockIdx.x * 32;
  const _Float16* kg = kb + (size_t)g * 65536;
  const _Float16* qg = qb + (size_t)g * 65536;
  v16h ak[2][2];
#pragma unroll
  for (int mt = 0; mt < 2; ++mt)
#pragma unroll
    for (int ks = 0; ks < 2; ++ks)
      ak[mt][ks] = frag_a_f16(kg + (size_t)(i0 + mt * 16) * 64, 64, lane, ks * 32);
  float mr[16], lr[16];
#pragma unroll
  for (int i = 0; i < 16; ++i) { mr[i] = -3.0e38f; lr[i] = 0.f; }
  for (int j0 = 0; j0 < 1024; j0 += 32) {
    v16h bq[2][2];
#pragma unroll
    for (int nt = 0; nt < 2; ++nt)
#pragma unroll
      for (int ks = 0; ks < 2; ++ks)
        bq[nt][ks] = frag_b_nm_f16(qg + (size_t)(j0 + nt * 16) * 64, 64, lane, ks * 32);
    v8f s[2][2] = {};
#pragma unroll
    for (int mt = 0; mt < 2; ++mt)
#pragma unroll
      for (int nt = 0; nt < 2; ++nt) {
        s[mt][nt] = wmma16(ak[mt][0], bq[nt][0], s[mt][nt]);
        s[mt][nt] = wmma16(ak[mt][1], bq[nt][1], s[mt][nt]);
      }
#pragma unroll
    for (int mt = 0; mt < 2; ++mt)
#pragma unroll
      for (int r8 = 0; r8 < 8; ++r8) {
        int idx = mt * 8 + r8;
        float mx = hmax16(fmaxf(s[mt][0][r8], s[mt][1][r8]));
        float nm = fmaxf(mr[idx], mx);
        float se = hsum16(__expf(s[mt][0][r8] - nm) + __expf(s[mt][1][r8] - nm));
        lr[idx] = lr[idx] * __expf(mr[idx] - nm) + se;
        mr[idx] = nm;
      }
  }
  if (ln == 0) {
#pragma unroll
    for (int mt = 0; mt < 2; ++mt)
#pragma unroll
      for (int r8 = 0; r8 < 8; ++r8) {
        int r = i0 + mt * 16 + sel * 8 + r8;
        rowmax[(size_t)g * 1024 + r] = mr[mt * 8 + r8];
        rowsum[(size_t)g * 1024 + r] = lr[mt * 8 + r8];
      }
  }
}

// ---- 4) attention pass2 (coarse): P.V + per-i-block column-sum partials ----
__global__ __launch_bounds__(32)
void k_attn_pass2_coarse(const _Float16* __restrict__ kb, const _Float16* __restrict__ qb,
                         const _Float16* __restrict__ vb,
                         const float* __restrict__ rowmax, const float* __restrict__ rowsum,
                         float* __restrict__ score_part, _Float16* __restrict__ anh) {
  __shared__ _Float16 pls[32][40];
  const int lane = threadIdx.x, sel = lane >> 4, ln = lane & 15;
  const int g = blockIdx.y, ib = blockIdx.x, i0 = ib * 32;
  const _Float16* kg = kb + (size_t)g * 65536;
  const _Float16* qg = qb + (size_t)g * 65536;
  const _Float16* vg = vb + (size_t)g * 65536;
  v16h ak[2][2];
#pragma unroll
  for (int mt = 0; mt < 2; ++mt)
#pragma unroll
    for (int ks = 0; ks < 2; ++ks)
      ak[mt][ks] = frag_a_f16(kg + (size_t)(i0 + mt * 16) * 64, 64, lane, ks * 32);
  float mrw[16], invl[16];
#pragma unroll
  for (int mt = 0; mt < 2; ++mt)
#pragma unroll
    for (int r8 = 0; r8 < 8; ++r8) {
      int r = i0 + mt * 16 + sel * 8 + r8;
      mrw[mt * 8 + r8] = rowmax[(size_t)g * 1024 + r];
      invl[mt * 8 + r8] = 1.f / rowsum[(size_t)g * 1024 + r];
    }
  v8f o[2][4] = {};
  for (int j0 = 0; j0 < 1024; j0 += 32) {
    v16h bq[2][2];
#pragma unroll
    for (int nt = 0; nt < 2; ++nt)
#pragma unroll
      for (int ks = 0; ks < 2; ++ks)
        bq[nt][ks] = frag_b_nm_f16(qg + (size_t)(j0 + nt * 16) * 64, 64, lane, ks * 32);
    v8f s[2][2] = {};
#pragma unroll
    for (int mt = 0; mt < 2; ++mt)
#pragma unroll
      for (int nt = 0; nt < 2; ++nt) {
        s[mt][nt] = wmma16(ak[mt][0], bq[nt][0], s[mt][nt]);
        s[mt][nt] = wmma16(ak[mt][1], bq[nt][1], s[mt][nt]);
      }
    __syncthreads();
#pragma unroll
    for (int nt = 0; nt < 2; ++nt) {
      float cs = 0.f;
#pragma unroll
      for (int mt = 0; mt < 2; ++mt)
#pragma unroll
        for (int r8 = 0; r8 < 8; ++r8) {
          float pv = __expf(s[mt][nt][r8] - mrw[mt * 8 + r8]) * invl[mt * 8 + r8];
          cs += pv;
          pls[mt * 16 + sel * 8 + r8][nt * 16 + ln] = (_Float16)pv;
        }
      cs += __shfl_xor(cs, 16, 32);
      if (lane < 16)
        score_part[(size_t)ib * 32768 + (size_t)g * 1024 + j0 + nt * 16 + ln] = cs;
    }
    __syncthreads();
    v16h pa[2];
#pragma unroll
    for (int mt = 0; mt < 2; ++mt)
      pa[mt] = frag_a_f16(&pls[mt * 16][0], 40, lane, 0);
#pragma unroll
    for (int nd = 0; nd < 4; ++nd) {
      v16h bv = frag_b_rm_f16(vg + (size_t)j0 * 64 + nd * 16, 64, lane, 0);
#pragma unroll
      for (int mt = 0; mt < 2; ++mt)
        o[mt][nd] = wmma16(pa[mt], bv, o[mt][nd]);
    }
  }
  const int b = g >> 2, head = g & 3;
#pragma unroll
  for (int mt = 0; mt < 2; ++mt)
#pragma unroll
    for (int nd = 0; nd < 4; ++nd)
#pragma unroll
      for (int r8 = 0; r8 < 8; ++r8) {
        int r = i0 + mt * 16 + sel * 8 + r8;
        int c = head * 64 + nd * 16 + ln;
        anh[((size_t)b * 1024 + r) * 256 + c] = (_Float16)o[mt][nd][r8];
      }
}

// ---- pass2 (topk variant): P.V, scatter-add into y at patch positions ----
__global__ __launch_bounds__(32)
void k_attn_pass2_topk(const _Float16* __restrict__ kb, const _Float16* __restrict__ qb,
                       const _Float16* __restrict__ vb,
                       const float* __restrict__ rowmax, const float* __restrict__ rowsum,
                       const int* __restrict__ topk, float* __restrict__ y) {
  __shared__ _Float16 pls[32][40];
  const int lane = threadIdx.x, sel = lane >> 4, ln = lane & 15;
  const int g = blockIdx.y, i0 = blockIdx.x * 32;
  const _Float16* kg = kb + (size_t)g * 65536;
  const _Float16* qg = qb + (size_t)g * 65536;
  const _Float16* vg = vb + (size_t)g * 65536;
  v16h ak[2][2];
#pragma unroll
  for (int mt = 0; mt < 2; ++mt)
#pragma unroll
    for (int ks = 0; ks < 2; ++ks)
      ak[mt][ks] = frag_a_f16(kg + (size_t)(i0 + mt * 16) * 64, 64, lane, ks * 32);
  float mrw[16], invl[16];
#pragma unroll
  for (int mt = 0; mt < 2; ++mt)
#pragma unroll
    for (int r8 = 0; r8 < 8; ++r8) {
      int r = i0 + mt * 16 + sel * 8 + r8;
      mrw[mt * 8 + r8] = rowmax[(size_t)g * 1024 + r];
      invl[mt * 8 + r8] = 1.f / rowsum[(size_t)g * 1024 + r];
    }
  v8f o[2][4] = {};
  for (int j0 = 0; j0 < 1024; j0 += 32) {
    v16h bq[2][2];
#pragma unroll
    for (int nt = 0; nt < 2; ++nt)
#pragma unroll
      for (int ks = 0; ks < 2; ++ks)
        bq[nt][ks] = frag_b_nm_f16(qg + (size_t)(j0 + nt * 16) * 64, 64, lane, ks * 32);
    v8f s[2][2] = {};
#pragma unroll
    for (int mt = 0; mt < 2; ++mt)
#pragma unroll
      for (int nt = 0; nt < 2; ++nt) {
        s[mt][nt] = wmma16(ak[mt][0], bq[nt][0], s[mt][nt]);
        s[mt][nt] = wmma16(ak[mt][1], bq[nt][1], s[mt][nt]);
      }
    __syncthreads();
#pragma unroll
    for (int nt = 0; nt < 2; ++nt)
#pragma unroll
      for (int mt = 0; mt < 2; ++mt)
#pragma unroll
        for (int r8 = 0; r8 < 8; ++r8) {
          float pv = __expf(s[mt][nt][r8] - mrw[mt * 8 + r8]) * invl[mt * 8 + r8];
          pls[mt * 16 + sel * 8 + r8][nt * 16 + ln] = (_Float16)pv;
        }
    __syncthreads();
    v16h pa[2];
#pragma unroll
    for (int mt = 0; mt < 2; ++mt)
      pa[mt] = frag_a_f16(&pls[mt * 16][0], 40, lane, 0);
#pragma unroll
    for (int nd = 0; nd < 4; ++nd) {
      v16h bv = frag_b_rm_f16(vg + (size_t)j0 * 64 + nd * 16, 64, lane, 0);
#pragma unroll
      for (int mt = 0; mt < 2; ++mt)
        o[mt][nd] = wmma16(pa[mt], bv, o[mt][nd]);
    }
  }
  const int b = g >> 2, head = g & 3;
#pragma unroll
  for (int mt = 0; mt < 2; ++mt)
#pragma unroll
    for (int nd = 0; nd < 4; ++nd)
#pragma unroll
      for (int r8 = 0; r8 < 8; ++r8) {
        int t = i0 + mt * 16 + sel * 8 + r8;
        int d = nd * 16 + ln;
        int kidx = t >> 2, dy = (t >> 1) & 1, dx = t & 1;
        int p = topk[g * 256 + kidx];
        int flat = p * 4 + dy * 2 + dx;           // torch row-major reshape
        int Y = flat >> 6, X = flat & 63;
        int c = head * 64 + d;
        size_t yi = (((size_t)b * 256 + c) * 64 + Y) * 64 + X;
        y[yi] += o[mt][nd][r8];                    // unique target per (t,d)
      }
}

// ---- score partial reduction (deterministic, no atomics) ----
__global__ void k_score_reduce(const float* __restrict__ part, float* __restrict__ score) {
  int i = blockIdx.x * blockDim.x + threadIdx.x;
  if (i < 32768) {
    float s = 0.f;
    for (int ib = 0; ib < 32; ++ib) s += part[(size_t)ib * 32768 + i];
    score[i] = s;
  }
}

// ---- 5) exact top-K=256 by rank counting (first-index tiebreak like lax.top_k) ----
__global__ __launch_bounds__(256)
void k_topk(const float* __restrict__ score, int* __restrict__ topk) {
  __shared__ float sc[1024];
  int g = blockIdx.x, tid = threadIdx.x;
  for (int i = tid; i < 1024; i += 256) sc[i] = score[(size_t)g * 1024 + i];
  __syncthreads();
  for (int r = 0; r < 4; ++r) {
    int i = tid + r * 256;
    float s = sc[i];
    int cnt = 0;
    for (int j = 0; j < 1024; ++j) {
      float t = sc[j];
      cnt += (t > s) || (t == s && j < i);
    }
    if (cnt < 256) topk[g * 256 + cnt] = i;
  }
}

// ---- 6) conv-transpose 4x4 s2 p1 as 4 parity implicit GEMMs: M=8192 N=256 K=1024 ----
__global__ __launch_bounds__(32)
void k_up(const _Float16* __restrict__ anh, const _Float16* __restrict__ wu,
          const float* __restrict__ bias, float* __restrict__ co) {
  const int lane = threadIdx.x, sel = lane >> 4, ln = lane & 15;
  const int m0 = blockIdx.x * 32, n0 = blockIdx.y * 32;
  const int par = blockIdx.z, py = par >> 1, px = par & 1;
  v8f acc[2][2] = {};
  int ob[2], oy2[2], ox2[2];
#pragma unroll
  for (int mt = 0; mt < 2; ++mt) {
    int m = m0 + mt * 16 + ln;
    ob[mt] = m >> 10; int q = m & 1023; oy2[mt] = q >> 5; ox2[mt] = q & 31;
  }
  for (int k0 = 0; k0 < 1024; k0 += 32) {
    int tap = k0 >> 8, ty = tap >> 1, tx = tap & 1;
    int ky = ((py ^ 1) & 1) + 2 * ty;
    int kx = ((px ^ 1) & 1) + 2 * tx;
    v16h a[2];
#pragma unroll
    for (int mt = 0; mt < 2; ++mt) {
      int iy = (2 * oy2[mt] + py + 1 - ky) / 2;   // always even numerator
      int ix = (2 * ox2[mt] + px + 1 - kx) / 2;
      bool ok = (unsigned)iy < 32u && (unsigned)ix < 32u;
      const _Float16* ap = anh + (((size_t)ob[mt] * 32 + iy) * 32 + ix) * 256 +
                           (k0 & 255) + sel * 16;
      if (ok) {
#pragma unroll
        for (int e = 0; e < 16; ++e) a[mt][e] = ap[e];
      } else {
#pragma unroll
        for (int e = 0; e < 16; ++e) a[mt][e] = (_Float16)0.f;
      }
    }
    v16h bf[2];
#pragma unroll
    for (int nt = 0; nt < 2; ++nt) {
      int n = n0 + nt * 16 + ln;
      const _Float16* wp = wu + (size_t)n * 4096 + (size_t)(ky * 4 + kx) * 256 +
                           (k0 & 255) + sel * 16;
#pragma unroll
      for (int e = 0; e < 16; ++e) bf[nt][e] = wp[e];
    }
#pragma unroll
    for (int mt = 0; mt < 2; ++mt)
#pragma unroll
      for (int nt = 0; nt < 2; ++nt)
        acc[mt][nt] = wmma16(a[mt], bf[nt], acc[mt][nt]);
  }
#pragma unroll
  for (int mt = 0; mt < 2; ++mt)
#pragma unroll
    for (int nt = 0; nt < 2; ++nt)
#pragma unroll
      for (int r8 = 0; r8 < 8; ++r8) {
        int m = m0 + mt * 16 + sel * 8 + r8;
        int n = n0 + nt * 16 + ln;
        int b = m >> 10, q = m & 1023;
        int oy = 2 * (q >> 5) + py, ox = 2 * (q & 31) + px;
        co[(((size_t)b * 256 + n) * 64 + oy) * 64 + ox] = acc[mt][nt][r8] + bias[n];
      }
}

// ---- 7) y = coarse_out + torch-reshape-scrambled(coarse_out) ----
__global__ void k_y_init(const float* __restrict__ co, float* __restrict__ y) {
  size_t i = (size_t)blockIdx.x * blockDim.x + threadIdx.x;
  if (i >= (size_t)8 * 256 * 4096) return;
  int flat = (int)(i & 4095);
  size_t bc = i >> 12;
  int p = flat >> 2, dy = (flat >> 1) & 1, dx = flat & 1;
  int Ys = 2 * (p >> 5) + dy, Xs = 2 * (p & 31) + dx;
  y[i] = co[i] + co[(bc << 12) + Ys * 64 + Xs];
}

// ---- 8) gather top-k 2x2 patches into f16 tokens ----
__global__ void k_gather(const float* __restrict__ co, const int* __restrict__ topk,
                         _Float16* __restrict__ tok2) {
  size_t i = (size_t)blockIdx.x * blockDim.x + threadIdx.x;
  if (i >= (size_t)32 * 1024 * 64) return;
  int d = (int)(i & 63);
  int t = (int)((i >> 6) & 1023);
  int g = (int)(i >> 16);
  int b = g >> 2, head = g & 3;
  int kidx = t >> 2, dy = (t >> 1) & 1, dx = t & 1;
  int p = topk[g * 256 + kidx];
  int Y = 2 * (p >> 5) + dy, X = 2 * (p & 31) + dx;
  int c = head * 64 + d;
  tok2[i] = (_Float16)co[(((size_t)b * 256 + c) * 64 + Y) * 64 + X];
}

// ---- 9) depthwise 3x3 + BN1 + ReLU6 -> f16 NHWC for the pointwise GEMM ----
__global__ void k_dw(const float* __restrict__ y, const float* __restrict__ w,
                     const float* __restrict__ g1, const float* __restrict__ b1,
                     const float* __restrict__ m1, const float* __restrict__ v1,
                     _Float16* __restrict__ y2) {
  size_t i = (size_t)blockIdx.x * blockDim.x + threadIdx.x;
  if (i >= (size_t)8 * 256 * 4096) return;
  int X = (int)(i & 63), Y = (int)((i >> 6) & 63);
  int c = (int)((i >> 12) & 255), b = (int)(i >> 20);
  const float* base = y + (((size_t)b * 256 + c) << 12);
  float s = 0.f;
#pragma unroll
  for (int ky = 0; ky < 3; ++ky)
#pragma unroll
    for (int kx = 0; kx < 3; ++kx) {
      int iy = Y + ky - 1, ix = X + kx - 1;
      if ((unsigned)iy < 64u && (unsigned)ix < 64u)
        s += w[c * 9 + ky * 3 + kx] * base[iy * 64 + ix];
    }
  float sc = g1[c] * rsqrtf(v1[c] + 1e-5f);
  float val = fminf(fmaxf(s * sc + (b1[c] - m1[c] * sc), 0.f), 6.f);
  y2[(((size_t)b * 64 + Y) * 64 + X) * 256 + c] = (_Float16)val;
}

// ---- 10) pointwise 1x1 GEMM M=32768 N=256 K=256 + BN2 + ReLU6 -> d_out ----
__global__ __launch_bounds__(32)
void k_pw(const _Float16* __restrict__ a, const _Float16* __restrict__ w,
          const float* __restrict__ g2, const float* __restrict__ b2,
          const float* __restrict__ m2, const float* __restrict__ v2,
          float* __restrict__ out) {
  const int lane = threadIdx.x, sel = lane >> 4, ln = lane & 15;
  const int m0 = blockIdx.x * 32, n0 = blockIdx.y * 32;
  v8f acc[2][2] = {};
  for (int k0 = 0; k0 < 256; k0 += 32) {
    v16h af[2], bf[2];
#pragma unroll
    for (int mt = 0; mt < 2; ++mt)
      af[mt] = frag_a_f16(a + (size_t)(m0 + mt * 16) * 256, 256, lane, k0);
#pragma unroll
    for (int nt = 0; nt < 2; ++nt)
      bf[nt] = frag_b_nm_f16(w + (size_t)(n0 + nt * 16) * 256, 256, lane, k0);
#pragma unroll
    for (int mt = 0; mt < 2; ++mt)
#pragma unroll
      for (int nt = 0; nt < 2; ++nt)
        acc[mt][nt] = wmma16(af[mt], bf[nt], acc[mt][nt]);
  }
#pragma unroll
  for (int mt = 0; mt < 2; ++mt)
#pragma unroll
    for (int nt = 0; nt < 2; ++nt) {
      int n = n0 + nt * 16 + ln;
      float sc = g2[n] * rsqrtf(v2[n] + 1e-5f);
      float sh = b2[n] - m2[n] * sc;
#pragma unroll
      for (int r8 = 0; r8 < 8; ++r8) {
        int m = m0 + mt * 16 + sel * 8 + r8;
        int b = m >> 12, flat = m & 4095;
        float val = fminf(fmaxf(acc[mt][nt][r8] * sc + sh, 0.f), 6.f);
        out[(((size_t)b * 256 + n) << 12) + flat] = val;
      }
    }
}

// ================= host launch =================
extern "C" void kernel_launch(void* const* d_in, const int* in_sizes, int n_in,
                              void* d_out, int out_size, void* d_ws, size_t ws_size,
                              hipStream_t stream) {
  const float* x      = (const float*)d_in[0];
  const float* down_w = (const float*)d_in[1];
  const float* down_b = (const float*)d_in[2];
  const float* up_w   = (const float*)d_in[3];
  const float* up_b   = (const float*)d_in[4];
  const float* cqkv_w = (const float*)d_in[5];
  const float* cqkv_b = (const float*)d_in[6];
  const float* tqkv_w = (const float*)d_in[7];
  const float* tqkv_b = (const float*)d_in[8];
  const float* dw_w   = (const float*)d_in[9];
  const float* bn1_g  = (const float*)d_in[10];
  const float* bn1_b  = (const float*)d_in[11];
  const float* bn1_m  = (const float*)d_in[12];
  const float* bn1_v  = (const float*)d_in[13];
  const float* pw_w   = (const float*)d_in[14];
  const float* bn2_g  = (const float*)d_in[15];
  const float* bn2_b  = (const float*)d_in[16];
  const float* bn2_m  = (const float*)d_in[17];
  const float* bn2_v  = (const float*)d_in[18];
  float* out = (float*)d_out;
  (void)in_sizes; (void)n_in; (void)out_size; (void)ws_size;

  char* ws = (char*)d_ws;
  size_t off = 0;
  auto alloc = [&](size_t bytes) -> void* {
    void* p = ws + off;
    off += (bytes + 255) & ~(size_t)255;
    return p;
  };
  _Float16* xh     = (_Float16*)alloc((size_t)8388608 * 2); // x NHWC f16
  _Float16* wdn    = (_Float16*)alloc((size_t)1048576 * 2); // down_w [n][tap][cin] f16
  _Float16* wup    = (_Float16*)alloc((size_t)1048576 * 2); // up_w [n][tap][cin] f16
  _Float16* wcq    = (_Float16*)alloc((size_t)12288 * 2);   // coarse qkv w f16
  _Float16* wtq    = (_Float16*)alloc((size_t)12288 * 2);   // topk qkv w f16
  _Float16* wpw    = (_Float16*)alloc((size_t)65536 * 2);   // pw w f16
  _Float16* tok    = (_Float16*)alloc((size_t)2097152 * 2); // G*1024*64 f16
  _Float16* qb     = (_Float16*)alloc((size_t)2097152 * 2);
  _Float16* kb     = (_Float16*)alloc((size_t)2097152 * 2);
  _Float16* vb     = (_Float16*)alloc((size_t)2097152 * 2);
  float*    rowmax = (float*)alloc((size_t)32768 * 4);
  float*    rowsum = (float*)alloc((size_t)32768 * 4);
  float*    spart  = (float*)alloc((size_t)32 * 32768 * 4); // per-iblock colsum partials
  float*    score  = (float*)alloc((size_t)32768 * 4);
  _Float16* anh    = (_Float16*)alloc((size_t)2097152 * 2); // attn out, [B][32][32][C] f16
  float*    co     = (float*)alloc((size_t)8388608 * 4);    // coarse_out NCHW f32
  int*      topk   = (int*)alloc((size_t)8192 * 4);
  float*    ybuf   = (float*)alloc((size_t)8388608 * 4);    // residual accumulator f32
  _Float16* y2     = (_Float16*)alloc((size_t)8388608 * 2); // dw output NHWC f16

  dim3 w32(32);
  // One-time layout/precision prep (f32 -> f16, GEMM-friendly layouts)
  k_x_to_nhwc<<<32768, 256, 0, stream>>>(x, xh);
  k_w_down_pack<<<4096, 256, 0, stream>>>(down_w, wdn);
  k_w_up_pack<<<4096, 256, 0, stream>>>(up_w, wup);
  k_w_cvt<<<48, 256, 0, stream>>>(cqkv_w, wcq, 12288);
  k_w_cvt<<<48, 256, 0, stream>>>(tqkv_w, wtq, 12288);
  k_w_cvt<<<256, 256, 0, stream>>>(pw_w, wpw, 65536);
  // Coarse branch
  k_down<<<dim3(256, 8), w32, 0, stream>>>(xh, wdn, down_b, tok);
  k_qkv<<<dim3(1024, 6), w32, 0, stream>>>(tok, wcq, cqkv_b, qb, kb, vb);
  k_attn_pass1<<<dim3(32, 32), w32, 0, stream>>>(kb, qb, rowmax, rowsum);
  k_attn_pass2_coarse<<<dim3(32, 32), w32, 0, stream>>>(kb, qb, vb, rowmax, rowsum,
                                                        spart, anh);
  k_score_reduce<<<128, 256, 0, stream>>>(spart, score);
  k_topk<<<32, 256, 0, stream>>>(score, topk);
  k_up<<<dim3(256, 8, 4), w32, 0, stream>>>(anh, wup, up_b, co);
  // Residual + top-k branch (reuse tok/qb/kb/vb/rowmax/rowsum)
  k_y_init<<<32768, 256, 0, stream>>>(co, ybuf);
  k_gather<<<8192, 256, 0, stream>>>(co, topk, tok);
  k_qkv<<<dim3(1024, 6), w32, 0, stream>>>(tok, wtq, tqkv_b, qb, kb, vb);
  k_attn_pass1<<<dim3(32, 32), w32, 0, stream>>>(kb, qb, rowmax, rowsum);
  k_attn_pass2_topk<<<dim3(32, 32), w32, 0, stream>>>(kb, qb, vb, rowmax, rowsum,
                                                      topk, ybuf);
  // DWConv block
  k_dw<<<32768, 256, 0, stream>>>(ybuf, dw_w, bn1_g, bn1_b, bn1_m, bn1_v, y2);
  k_pw<<<dim3(1024, 8), w32, 0, stream>>>(y2, wpw, bn2_g, bn2_b, bn2_m, bn2_v, out);
}
